// SGSL_10797547782573
// MI455X (gfx1250) — matrix operand order repaired and checked by reading, not compiled, not run on hardware
//
#include <hip/hip_runtime.h>
#include <hip/hip_bf16.h>

// ---------------- problem constants (match reference) ----------------
#define NUc 50000
#define NFc 50000
#define Nc  100000      // NU + NF
#define Dc  64
#define H2c 32
#define FUc 256
#define Ec  500000
#define EPc 500000
#define ENc 500000
#define THRESHc 0.3f

typedef __attribute__((ext_vector_type(16))) _Float16 v16h;
typedef __attribute__((ext_vector_type(8)))  float    v8f;

static inline int cdiv(long long a, long long b) { return (int)((a + b - 1) / b); }

__device__ inline void atomAddF(float* p, float v) { unsafeAtomicAdd(p, v); }

__device__ inline float waveReduceSum(float v) {
  #pragma unroll
  for (int off = 16; off > 0; off >>= 1) v += __shfl_down(v, off, 32);
  return v;
}

// =====================================================================
// WMMA GEMM: C[M x NOUT] = act(A[M x K] @ W[K x NOUT] + bias)
// A,W row-major f32, converted to f16 in LDS.  NOUT = NTILES*16.
// One block = 128 threads = 4 waves, each wave owns 16 rows of M.
// W is stored transposed in LDS ([n][k]) so B fragments are contiguous.
// =====================================================================
template <int K, int NTILES>
__global__ __launch_bounds__(128) void gemm_wmma_kern(
    const float* __restrict__ A, const float* __restrict__ W,
    const float* __restrict__ bias, float* __restrict__ C,
    int M, int ldc, int colOff, int doRelu)
{
  constexpr int NOUT = NTILES * 16;
  __shared__ _Float16 Wt[NOUT * K];   // [n][k]
  __shared__ _Float16 At[64 * K];     // [r][k], 4 waves x 16 rows

  const int tid = threadIdx.x;
  const int rowBase = blockIdx.x * 64;

  for (int idx = tid; idx < K * NOUT; idx += 128) {
    int k = idx / NOUT, n = idx - k * NOUT;
    Wt[n * K + k] = (_Float16)W[idx];
  }
  for (int idx = tid; idx < 64 * K; idx += 128) {
    int r = idx / K, k = idx - r * K;
    int row = rowBase + r;
    At[idx] = (_Float16)(row < M ? A[(size_t)row * K + k] : 0.0f);
  }
  __syncthreads();

  const int wave = tid >> 5, lane = tid & 31;
  const int half = lane >> 4, n16 = lane & 15, mloc = lane & 15;

  v8f zero = {0.f, 0.f, 0.f, 0.f, 0.f, 0.f, 0.f, 0.f};
  v8f acc[NTILES];
  #pragma unroll
  for (int t = 0; t < NTILES; ++t) acc[t] = zero;

  const _Float16* Arow = &At[(wave * 16 + mloc) * K];

  #pragma unroll
  for (int kb = 0; kb < K; kb += 32) {
    // A fragment: 16x32 f16 layout (ISA 7.12.2)
    v16h a;
    #pragma unroll
    for (int j = 0; j < 8; ++j) {
      int kidx = (j < 4) ? (kb + half * 8 + 2 * j)
                         : (kb + 16 + half * 8 + 2 * (j - 4));
      a[2 * j]     = Arow[kidx];
      a[2 * j + 1] = Arow[kidx + 1];
    }
    #pragma unroll
    for (int t = 0; t < NTILES; ++t) {
      // B fragment: 32x16, lanes 0-15 hold K=kb..kb+7,kb+16..kb+23, lanes 16-31 K=kb+8..kb+15,kb+24..kb+31
      const _Float16* Wrow = &Wt[(t * 16 + n16) * K + kb + half * 8];
      v16h b;
      #pragma unroll
      for (int j = 0; j < 16; ++j) b[j] = Wrow[(j < 8) ? j : j + 8];
      acc[t] = __builtin_amdgcn_wmma_f32_16x16x32_f16(
          false, a, false, b, (short)0, acc[t], false, false);
    }
  }

  #pragma unroll
  for (int t = 0; t < NTILES; ++t) {
    int col = colOff + t * 16 + n16;
    float bv = bias[t * 16 + n16];
    #pragma unroll
    for (int r = 0; r < 8; ++r) {
      int row = rowBase + wave * 16 + half * 8 + r;
      if (row < M) {
        float v = acc[t][r] + bv;
        if (doRelu) v = fmaxf(v, 0.0f);
        C[(size_t)row * ldc + col] = v;
      }
    }
  }
}

// ---------------- elementwise / gather-scatter helpers ----------------

// O = concat(Xa[0:na], Xb[0:nb])
__global__ void cat2_kern(const float* __restrict__ Xa, long long na,
                          const float* __restrict__ Xb, long long nb,
                          float* __restrict__ O)
{
  long long i = (long long)blockIdx.x * blockDim.x + threadIdx.x;
  if (i >= na + nb) return;
  O[i] = (i < na) ? Xa[i] : Xb[i - na];
}

__global__ void count_kern(const int* __restrict__ dst, int nE, float* __restrict__ cnt)
{
  int e = blockIdx.x * blockDim.x + threadIdx.x;
  if (e >= nE) return;
  atomAddF(&cnt[dst[e]], 1.0f);
}

// one wave per edge: S[dst][0:ncols] += X[src][xOff:xOff+ncols]
__global__ void scatter_add_kern(const float* __restrict__ X, int ldx, int xOff, int ncols,
                                 const int* __restrict__ src, const int* __restrict__ dst,
                                 int nE, float* __restrict__ S)
{
  int gid = blockIdx.x * blockDim.x + threadIdx.x;
  int e = gid >> 5, lane = gid & 31;
  if (e >= nE) return;
  const float* xr = X + (size_t)src[e] * ldx + xOff;
  float* sr = S + (size_t)dst[e] * ncols;
  for (int c = lane; c < ncols; c += 32) atomAddF(&sr[c], xr[c]);
}

__global__ void mean_div_kern(const float* __restrict__ S, const float* __restrict__ cnt,
                              int n, int ncols, float* __restrict__ O, int ldo, int oOff)
{
  long long idx = (long long)blockIdx.x * blockDim.x + threadIdx.x;
  if (idx >= (long long)n * ncols) return;
  int i = (int)(idx / ncols), c = (int)(idx - (long long)i * ncols);
  O[(size_t)i * ldo + oOff + c] = S[idx] / fmaxf(cnt[i], 1.0f);
}

__global__ void copy_cols_kern(const float* __restrict__ X, int ldx, int xOff, int ncols,
                               int n, float* __restrict__ O, int ldo, int oOff)
{
  long long idx = (long long)blockIdx.x * blockDim.x + threadIdx.x;
  if (idx >= (long long)n * ncols) return;
  int i = (int)(idx / ncols), c = (int)(idx - (long long)i * ncols);
  O[(size_t)i * ldo + oOff + c] = X[(size_t)i * ldx + xOff + c];
}

// ---------------- per-edge kernels ----------------

__global__ void edge_sim_kern(const float* __restrict__ u, const float* __restrict__ f,
                              const int* __restrict__ e0, const int* __restrict__ e1,
                              int nE, const float* __restrict__ metric_w,
                              float* __restrict__ mask_f, float* __restrict__ sumAbs)
{
  __shared__ float mw[4 * 64];
  for (int i = threadIdx.x; i < 256; i += blockDim.x) mw[i] = metric_w[i];
  __syncthreads();
  int e = blockIdx.x * blockDim.x + threadIdx.x;
  float m = 0.0f;
  if (e < nE) {
    const float* ur = u + (size_t)e0[e] * 64;
    const float* fr = f + (size_t)e1[e] * 64;
    float dt[4] = {0, 0, 0, 0}, na[4] = {0, 0, 0, 0}, nb[4] = {0, 0, 0, 0};
    for (int d = 0; d < 64; ++d) {
      float ud = ur[d], fd = fr[d];
      #pragma unroll
      for (int h = 0; h < 4; ++h) {
        float w = mw[h * 64 + d];
        float a = ud * w, b = fd * w;
        dt[h] += a * b; na[h] += a * a; nb[h] += b * b;
      }
    }
    float sim = 0.0f;
    #pragma unroll
    for (int h = 0; h < 4; ++h)
      sim += dt[h] / (fmaxf(sqrtf(na[h]), 1e-8f) * fmaxf(sqrtf(nb[h]), 1e-8f));
    sim *= 0.25f;
    m = (sim < THRESHc) ? 0.0f : sim;
    mask_f[e] = m;
  }
  float s = waveReduceSum(fabsf(m));
  if ((threadIdx.x & 31) == 0) atomicAdd(sumAbs, s);
}

__global__ void edge_semantic_kern(const float* __restrict__ z,
                                   const int* __restrict__ e0, const int* __restrict__ e1,
                                   int nE, const float* __restrict__ lin_w,
                                   const float* __restrict__ lin_b,
                                   float* __restrict__ mask_s, float* __restrict__ sumAbs)
{
  __shared__ float lw[128 * 3];
  __shared__ float lb[3];
  for (int i = threadIdx.x; i < 384; i += blockDim.x) lw[i] = lin_w[i];
  if (threadIdx.x < 3) lb[threadIdx.x] = lin_b[threadIdx.x];
  __syncthreads();
  int e = blockIdx.x * blockDim.x + threadIdx.x;
  float m = 0.0f;
  if (e < nE) {
    const float* z0 = z + (size_t)e0[e] * 64;
    const float* z1 = z + (size_t)e1[e] * 64;   // raw e1 (matches reference)
    float v0 = lb[0], v1 = lb[1], v2 = lb[2];
    for (int d = 0; d < 64; ++d) {
      float a = z0[d];
      v0 += a * lw[d * 3 + 0]; v1 += a * lw[d * 3 + 1]; v2 += a * lw[d * 3 + 2];
    }
    for (int d = 0; d < 64; ++d) {
      float a = z1[d];
      v0 += a * lw[(64 + d) * 3 + 0]; v1 += a * lw[(64 + d) * 3 + 1]; v2 += a * lw[(64 + d) * 3 + 2];
    }
    int best = 0; float bv = v0;                 // argmax, first-occurrence ties
    if (v1 > bv) { bv = v1; best = 1; }
    if (v2 > bv) { bv = v2; best = 2; }
    m = (float)best - 1.0f;                      // [-1,0,1][cls]
    mask_s[e] = m;
  }
  float s = waveReduceSum(fabsf(m));
  if ((threadIdx.x & 31) == 0) atomicAdd(sumAbs, s);
}

__global__ void fuse_kern(const float* __restrict__ mask_f, const float* __restrict__ mask_s,
                          const float* __restrict__ fusion_w, const float* __restrict__ sumAbs,
                          const int* __restrict__ e0, int nE, float Etot,
                          float* __restrict__ w_edge, float* __restrict__ deg)
{
  int e = blockIdx.x * blockDim.x + threadIdx.x;
  if (e >= nE) return;
  float f0 = fusion_w[0], f1 = fusion_w[1], f2 = fusion_w[2];
  float mx = fmaxf(f0, fmaxf(f1, f2));
  float x0 = __expf(f0 - mx), x1 = __expf(f1 - mx), x2 = __expf(f2 - mx);
  float inv = 1.0f / (x0 + x1 + x2);
  float w0 = x0 * inv, w1 = x1 * inv, w2 = x2 * inv;
  float sO = fmaxf(Etot, 1e-12f);
  float sF = fmaxf(sumAbs[0], 1e-12f);
  float sS = fmaxf(sumAbs[1], 1e-12f);
  float fused = w0 * (1.0f / sO) + w1 * (mask_f[e] / sF) + w2 * (mask_s[e] / sS);
  float we = (fused > 0.5f) ? 1.0f : 0.0f;
  w_edge[e] = we;
  if (we != 0.0f) atomAddF(&deg[e0[e]], we);
}

__global__ void node_dis_kern(const float* __restrict__ deg, int n, float* __restrict__ dis)
{
  int i = blockIdx.x * blockDim.x + threadIdx.x;
  if (i >= n) return;
  float d = deg[i];
  dis[i] = (d > 0.0f) ? rsqrtf(fmaxf(d, 1e-12f)) : 0.0f;
}

__global__ void edge_norm_kern(const float* __restrict__ dis, const float* __restrict__ w_edge,
                               const int* __restrict__ e0, const int* __restrict__ e1,
                               int nE, float* __restrict__ normv)
{
  int e = blockIdx.x * blockDim.x + threadIdx.x;
  if (e >= nE) return;
  normv[e] = dis[e0[e]] * w_edge[e] * dis[e1[e]];
}

// ---------------- LightGCN ----------------

__global__ void lgc_init_kern(const float* __restrict__ lgu, const float* __restrict__ lgi,
                              float* __restrict__ acc, float* __restrict__ xk)
{
  long long i = (long long)blockIdx.x * blockDim.x + threadIdx.x;
  if (i >= (long long)Nc * 64) return;
  float v = (i < (long long)NUc * 64) ? lgu[i] : lgi[i - (long long)NUc * 64];
  acc[i] = v; xk[i] = v;
}

__global__ void lgc_scatter_kern(const float* __restrict__ xk, const float* __restrict__ normv,
                                 const int* __restrict__ e0, const int* __restrict__ e1,
                                 int nE, float* __restrict__ xn)
{
  int gid = blockIdx.x * blockDim.x + threadIdx.x;
  int e = gid >> 5, lane = gid & 31;
  if (e >= nE) return;
  float nv = normv[e];
  if (nv == 0.0f) return;                        // wave-uniform skip
  const float* src = xk + (size_t)e1[e] * 64;    // raw e1 (matches reference)
  float* dst = xn + (size_t)e0[e] * 64;
  atomAddF(&dst[lane],      nv * src[lane]);
  atomAddF(&dst[lane + 32], nv * src[lane + 32]);
}

__global__ void lgc_acc_kern(float* __restrict__ acc, const float* __restrict__ xn, long long n)
{
  long long i = (long long)blockIdx.x * blockDim.x + threadIdx.x;
  if (i >= n) return;
  acc[i] += xn[i];
}

__global__ void write_out_kern(const float* __restrict__ acc, const float* __restrict__ lgu,
                               const float* __restrict__ lgi, float* __restrict__ out)
{
  long long i = (long long)blockIdx.x * blockDim.x + threadIdx.x;
  const long long S1 = (long long)NUc * 64;
  const long long S2 = 2 * S1;
  const long long S3 = S2 + (long long)NFc * 64;
  const long long TOT = S3 + (long long)NFc * 64;
  if (i >= TOT) return;
  if (i < S1)      out[i] = acc[i] * 0.25f;                 // final[:NU] = acc/(L+1)
  else if (i < S2) out[i] = lgu[i - S1];                    // lg_users_emb
  else if (i < S3) out[i] = acc[S1 + (i - S2)] * 0.25f;     // final[NU:]
  else             out[i] = lgi[i - S3];                    // lg_items_emb
}

// =====================================================================

extern "C" void kernel_launch(void* const* d_in, const int* in_sizes, int n_in,
                              void* d_out, int out_size, void* d_ws, size_t ws_size,
                              hipStream_t stream)
{
  (void)in_sizes; (void)n_in; (void)out_size; (void)ws_size;
  // ----- inputs (setup_inputs order) -----
  const float* user_feat = (const float*)d_in[0];
  const float* food_feat = (const float*)d_in[1];
  const int*   edge_i    = (const int*)d_in[2];     // (2,E)
  const int*   pos_i     = (const int*)d_in[3];     // (2,EP)
  const int*   neg_i     = (const int*)d_in[4];     // (2,EN)
  const float* W_user    = (const float*)d_in[5];
  const float* b_user    = (const float*)d_in[6];
  const float* W_food    = (const float*)d_in[7];
  const float* b_food    = (const float*)d_in[8];
  const float* metric_w  = (const float*)d_in[9];
  const float* fusion_w  = (const float*)d_in[10];
  const float* sg_users  = (const float*)d_in[11];
  const float* sg_items  = (const float*)d_in[12];
  const float* c1_pos_l  = (const float*)d_in[13];
  const float* c1_pos_rw = (const float*)d_in[14];
  const float* c1_pos_rb = (const float*)d_in[15];
  const float* c1_neg_l  = (const float*)d_in[16];
  const float* c1_neg_rw = (const float*)d_in[17];
  const float* c1_neg_rb = (const float*)d_in[18];
  const float* cs_pos_l  = (const float*)d_in[19];  // (2,64,32)
  const float* cs_pos_rw = (const float*)d_in[20];  // (2,32,32)
  const float* cs_pos_rb = (const float*)d_in[21];  // (2,32)
  const float* cs_neg_l  = (const float*)d_in[22];
  const float* cs_neg_rw = (const float*)d_in[23];
  const float* cs_neg_rb = (const float*)d_in[24];
  const float* lin_w     = (const float*)d_in[25];
  const float* lin_b     = (const float*)d_in[26];
  const float* lg_users  = (const float*)d_in[27];
  const float* lg_items  = (const float*)d_in[28];
  float* out = (float*)d_out;

  const int* e0  = edge_i;            const int* e1  = edge_i + Ec;
  const int* pe0 = pos_i;             const int* pe1 = pos_i + EPc;
  const int* ne0 = neg_i;             const int* ne1 = neg_i + ENc;

  // ----- workspace arena (floats) -----
  float* wsf = (float*)d_ws;
  size_t off = 0;
  auto alloc = [&](size_t n) { float* p = wsf + off; off += n; return p; };
  float* xbuf   = alloc((size_t)Nc * 64);
  float* u_out  = alloc((size_t)NUc * 64);
  float* f_out  = alloc((size_t)NFc * 64);
  float* mask_f = alloc(Ec);
  float* mask_s = alloc(Ec);
  float* scal   = alloc(16);
  float* cnt_p  = alloc(Nc);
  float* cnt_n  = alloc(Nc);
  float* sums   = alloc((size_t)Nc * 128);   // layer1: [sum_p | sum_n]; layers: s0..s3 (N*32)
  float* PQ     = alloc((size_t)Nc * 256);   // P,Q; later acc/xkA/xkB
  float* zb     = alloc((size_t)Nc * 64);
  float* z2b    = alloc((size_t)Nc * 64);
  float* wcat   = alloc(2 * 128 * 32 + 4 * 96 * 32);
  float* w_edge = alloc(Ec);
  float* deg    = alloc(Nc);
  float* disv   = alloc(Nc);
  float* normv  = alloc(Ec);

  float* sum_p = sums;
  float* sum_n = sums + (size_t)Nc * 64;
  float* P = PQ;
  float* Q = PQ + (size_t)Nc * 128;
  float* Wp1 = wcat;
  float* Wn1 = Wp1 + 128 * 32;
  float* Wp2 = Wn1 + 128 * 32;
  float* Wn2 = Wp2 + 96 * 32;
  float* Wp3 = Wn2 + 96 * 32;
  float* Wn3 = Wp3 + 96 * 32;

  const int T = 256;
  const long long NE64 = (long long)Nc * 64;

  // zero accumulators (graph-capture-safe)
  hipMemsetAsync(scal, 0, 16 * sizeof(float), stream);
  hipMemsetAsync(cnt_p, 0, Nc * sizeof(float), stream);
  hipMemsetAsync(cnt_n, 0, Nc * sizeof(float), stream);
  hipMemsetAsync(sums, 0, (size_t)Nc * 128 * sizeof(float), stream);
  hipMemsetAsync(deg, 0, Nc * sizeof(float), stream);

  // ---- 1. dense feature transforms (WMMA) ----
  gemm_wmma_kern<256, 4><<<cdiv(NUc, 64), 128, 0, stream>>>(user_feat, W_user, b_user, u_out, NUc, 64, 0, 1);
  gemm_wmma_kern<256, 4><<<cdiv(NFc, 64), 128, 0, stream>>>(food_feat, W_food, b_food, f_out, NFc, 64, 0, 1);

  // ---- 2. per-edge multi-head cosine similarity mask ----
  edge_sim_kern<<<cdiv(Ec, T), T, 0, stream>>>(u_out, f_out, e0, e1, Ec, metric_w, mask_f, &scal[0]);

  // ---- 3. signed graph conv ----
  cat2_kern<<<cdiv(NE64, T), T, 0, stream>>>(sg_users, (long long)NUc * 64, sg_items, (long long)NFc * 64, xbuf);
  count_kern<<<cdiv(EPc, T), T, 0, stream>>>(pe1, EPc, cnt_p);
  count_kern<<<cdiv(ENc, T), T, 0, stream>>>(ne1, ENc, cnt_n);

  // layer 1: seg-means of x over pos/neg edges
  scatter_add_kern<<<cdiv((long long)EPc * 32, T), T, 0, stream>>>(xbuf, 64, 0, 64, pe0, pe1, EPc, sum_p);
  scatter_add_kern<<<cdiv((long long)ENc * 32, T), T, 0, stream>>>(xbuf, 64, 0, 64, ne0, ne1, ENc, sum_n);
  mean_div_kern<<<cdiv(NE64, T), T, 0, stream>>>(sum_p, cnt_p, Nc, 64, P, 128, 0);
  copy_cols_kern<<<cdiv(NE64, T), T, 0, stream>>>(xbuf, 64, 0, 64, Nc, P, 128, 64);
  mean_div_kern<<<cdiv(NE64, T), T, 0, stream>>>(sum_n, cnt_n, Nc, 64, Q, 128, 0);
  copy_cols_kern<<<cdiv(NE64, T), T, 0, stream>>>(xbuf, 64, 0, 64, Nc, Q, 128, 64);

  // concatenated weight blocks
  cat2_kern<<<cdiv(128 * 32, T), T, 0, stream>>>(c1_pos_l, 64 * 32, c1_pos_rw, 64 * 32, Wp1);
  cat2_kern<<<cdiv(128 * 32, T), T, 0, stream>>>(c1_neg_l, 64 * 32, c1_neg_rw, 64 * 32, Wn1);
  cat2_kern<<<cdiv(96 * 32, T), T, 0, stream>>>(cs_pos_l,           64 * 32, cs_pos_rw,           32 * 32, Wp2);
  cat2_kern<<<cdiv(96 * 32, T), T, 0, stream>>>(cs_neg_l,           64 * 32, cs_neg_rw,           32 * 32, Wn2);
  cat2_kern<<<cdiv(96 * 32, T), T, 0, stream>>>(cs_pos_l + 64 * 32, 64 * 32, cs_pos_rw + 32 * 32, 32 * 32, Wp3);
  cat2_kern<<<cdiv(96 * 32, T), T, 0, stream>>>(cs_neg_l + 64 * 32, 64 * 32, cs_neg_rw + 32 * 32, 32 * 32, Wn3);

  // layer 1 GEMMs -> zb (pos cols 0-31, neg cols 32-63), relu
  gemm_wmma_kern<128, 2><<<cdiv(Nc, 64), 128, 0, stream>>>(P, Wp1, c1_pos_rb, zb, Nc, 64, 0, 1);
  gemm_wmma_kern<128, 2><<<cdiv(Nc, 64), 128, 0, stream>>>(Q, Wn1, c1_neg_rb, zb, Nc, 64, 32, 1);

  // layers 2..3
  for (int l = 0; l < 2; ++l) {
    float* zin  = (l == 0) ? zb : z2b;
    float* zout = (l == 0) ? z2b : zb;
    const float* Wp = (l == 0) ? Wp2 : Wp3;
    const float* Wn = (l == 0) ? Wn2 : Wn3;
    const float* bp = cs_pos_rb + l * 32;
    const float* bn = cs_neg_rb + l * 32;
    float* s0 = sums;
    float* s1 = sums + (size_t)Nc * 32;
    float* s2 = sums + (size_t)Nc * 64;
    float* s3 = sums + (size_t)Nc * 96;
    hipMemsetAsync(sums, 0, (size_t)Nc * 128 * sizeof(float), stream);
    scatter_add_kern<<<cdiv((long long)EPc * 32, T), T, 0, stream>>>(zin, 64, 0,  32, pe0, pe1, EPc, s0); // xp over pos
    scatter_add_kern<<<cdiv((long long)ENc * 32, T), T, 0, stream>>>(zin, 64, 32, 32, ne0, ne1, ENc, s1); // xn over neg
    scatter_add_kern<<<cdiv((long long)EPc * 32, T), T, 0, stream>>>(zin, 64, 32, 32, pe0, pe1, EPc, s2); // xn over pos
    scatter_add_kern<<<cdiv((long long)ENc * 32, T), T, 0, stream>>>(zin, 64, 0,  32, ne0, ne1, ENc, s3); // xp over neg
    long long N32 = (long long)Nc * 32;
    mean_div_kern<<<cdiv(N32, T), T, 0, stream>>>(s0, cnt_p, Nc, 32, P, 96, 0);
    mean_div_kern<<<cdiv(N32, T), T, 0, stream>>>(s1, cnt_n, Nc, 32, P, 96, 32);
    copy_cols_kern<<<cdiv(N32, T), T, 0, stream>>>(zin, 64, 0, 32, Nc, P, 96, 64);
    mean_div_kern<<<cdiv(N32, T), T, 0, stream>>>(s2, cnt_p, Nc, 32, Q, 96, 0);
    mean_div_kern<<<cdiv(N32, T), T, 0, stream>>>(s3, cnt_n, Nc, 32, Q, 96, 32);
    copy_cols_kern<<<cdiv(N32, T), T, 0, stream>>>(zin, 64, 32, 32, Nc, Q, 96, 64);
    gemm_wmma_kern<96, 2><<<cdiv(Nc, 64), 128, 0, stream>>>(P, Wp, bp, zout, Nc, 64, 0, 1);
    gemm_wmma_kern<96, 2><<<cdiv(Nc, 64), 128, 0, stream>>>(Q, Wn, bn, zout, Nc, 64, 32, 1);
  }
  // final z is in zb

  // ---- 4. semantic mask ----
  edge_semantic_kern<<<cdiv(Ec, T), T, 0, stream>>>(zb, e0, e1, Ec, lin_w, lin_b, mask_s, &scal[1]);

  // ---- 5. mask fusion -> edge weights -> degree norm ----
  fuse_kern<<<cdiv(Ec, T), T, 0, stream>>>(mask_f, mask_s, fusion_w, scal, e0, Ec, (float)Ec, w_edge, deg);
  node_dis_kern<<<cdiv(Nc, T), T, 0, stream>>>(deg, Nc, disv);
  edge_norm_kern<<<cdiv(Ec, T), T, 0, stream>>>(disv, w_edge, e0, e1, Ec, normv);

  // ---- 6. LightGCN (PQ region recycled: acc / xkA / xkB) ----
  float* acc = PQ;
  float* xkA = PQ + (size_t)Nc * 64;
  float* xkB = PQ + (size_t)Nc * 128;
  lgc_init_kern<<<cdiv(NE64, T), T, 0, stream>>>(lg_users, lg_items, acc, xkA);
  float* cur = xkA;
  float* nxt = xkB;
  for (int k = 0; k < 3; ++k) {
    hipMemsetAsync(nxt, 0, NE64 * sizeof(float), stream);
    lgc_scatter_kern<<<cdiv((long long)Ec * 32, T), T, 0, stream>>>(cur, normv, e0, e1, Ec, nxt);
    lgc_acc_kern<<<cdiv(NE64, T), T, 0, stream>>>(acc, nxt, NE64);
    float* tmp = cur; cur = nxt; nxt = tmp;
  }

  // ---- 7. outputs: (final[:NU], lg_users, final[NU:], lg_items) ----
  write_out_kern<<<cdiv(2 * NE64, T), T, 0, stream>>>(acc, lg_users, lg_items, out);
}
